// CitationGraphTransformer_59244778881418
// MI455X (gfx1250) — matrix operand order repaired and checked by reading, not compiled, not run on hardware
//
#include <hip/hip_runtime.h>
#include <math.h>

// ---------------- constants ----------------
#define Bb 128
#define Nn 512
#define TDd 768
#define Dd 512
#define HSs 256
#define Hh 8
#define HDd 64
#define NLl 4
#define FFf 2048
#define Kk 64
#define MDd 16
#define Ss 65            // K+1
#define NEGV (-1.0e9f)
#define SCALEV 0.125f    // 64^-0.5

// ---------------- optional gfx1250 async global->LDS ----------------
#if defined(__has_builtin)
#if __has_builtin(__builtin_amdgcn_global_load_async_to_lds_b128) && \
    __has_builtin(__builtin_amdgcn_s_wait_asynccnt)
#define HAVE_ASYNC_LDS 1
#endif
#endif

// ---------------- WMMA / native vector types ----------------
typedef __attribute__((ext_vector_type(16))) __bf16 v16bf;
typedef __attribute__((ext_vector_type(8)))  float  v8f;
typedef __attribute__((ext_vector_type(4)))  int    v4i;   // native int4 (clang vector)

union BF16Frag { v16bf v; v4i i4[2]; };

__device__ __forceinline__ unsigned short f2bf(float f) {
  unsigned int u = __float_as_uint(f);
  u += 0x7FFFu + ((u >> 16) & 1u);   // round to nearest even
  return (unsigned short)(u >> 16);
}

__device__ __forceinline__ unsigned pack_bf16(float lo, float hi) {
  return (unsigned)f2bf(lo) | ((unsigned)f2bf(hi) << 16);
}

__device__ __forceinline__ float gelu_f(float x) {
  float x3 = x * x * x;
  return 0.5f * x * (1.0f + tanhf(0.79788456080286535588f * (x + 0.044715f * x3)));
}

// ---------------- generic WMMA GEMM ----------------
// C[M,N] = act(A[M,K(lda)] @ W[K,N] + bias) (+ C if addC); ldc row stride of C.
// M,N multiples of 128; K arbitrary (zero-padded tail handled out of hot loop).
#define TM 128
#define TN 128
#define TK 32

__global__ __launch_bounds__(256) void wmma_gemm_kernel(
    const float* __restrict__ A, int lda,
    const float* __restrict__ W,
    const float* __restrict__ bias,
    float* __restrict__ C, int ldc,
    int M, int N, int K, int act, int addC)
{
  __shared__ __align__(16) unsigned short As[TM][TK];   // [m][k]
  __shared__ __align__(16) unsigned short Bs[TN][TK];   // [n][k] (W transposed)
  const int tid  = threadIdx.x;
  const int wave = tid >> 5;
  const int lane = tid & 31;
  const int m0 = blockIdx.y * TM;
  const int n0 = blockIdx.x * TN;
  const int wm = (wave >> 1) * 32;   // 0,32,64,96
  const int wn = (wave & 1) * 64;    // 0,64
  const int halfsel = lane >> 4;     // 0 or 1
  const int l16 = lane & 15;

  v8f acc[2][4];
  #pragma unroll
  for (int i = 0; i < 2; ++i)
    #pragma unroll
    for (int j = 0; j < 4; ++j)
      #pragma unroll
      for (int r = 0; r < 8; ++r) acc[i][j][r] = 0.0f;

  auto stage = [&](int k0, bool guard) {
    // A tile: 128 rows x 32 k -> 2048 packed u32; each thread packs 8
    #pragma unroll
    for (int it = 0; it < 8; ++it) {
      int idx = it * 256 + tid;
      int r = idx >> 4, kp = idx & 15;       // 16 k-pairs per row
      int gk = k0 + 2 * kp;
      const float* ap = A + (size_t)(m0 + r) * lda + gk;
      float v0, v1;
      if (!guard) { v0 = ap[0]; v1 = ap[1]; }
      else {
        v0 = (gk     < K) ? ap[0] : 0.0f;
        v1 = (gk + 1 < K) ? ap[1] : 0.0f;
      }
      *(unsigned*)&As[r][2 * kp] = pack_bf16(v0, v1);
    }
    // B tile: 32 k x 128 n, transposed into Bs[n][k]
    #pragma unroll
    for (int it = 0; it < 8; ++it) {
      int idx = it * 256 + tid;
      int n = idx & 127, kp = idx >> 7;      // kp 0..15
      int gk = k0 + 2 * kp;
      const float* wp = W + (size_t)gk * N + (n0 + n);
      float v0, v1;
      if (!guard) { v0 = wp[0]; v1 = wp[N]; }
      else {
        v0 = (gk     < K) ? wp[0] : 0.0f;
        v1 = (gk + 1 < K) ? wp[N] : 0.0f;
      }
      *(unsigned*)&Bs[n][2 * kp] = pack_bf16(v0, v1);
    }
  };

  auto compute = [&]() {
    BF16Frag a[2], b[4];
    #pragma unroll
    for (int i = 0; i < 2; ++i) {
      const v4i* p = (const v4i*)(&As[wm + i * 16 + l16][0]);
      a[i].i4[0] = p[halfsel];       // K = halfsel*8 .. +7
      a[i].i4[1] = p[2 + halfsel];   // K = 16 + halfsel*8 .. +7
    }
    #pragma unroll
    for (int j = 0; j < 4; ++j) {
      const v4i* p = (const v4i*)(&Bs[wn + j * 16 + l16][0]);
      b[j].i4[0] = p[halfsel];       // K = halfsel*8 .. +7
      b[j].i4[1] = p[2 + halfsel];   // K = 16 + halfsel*8 .. +7
    }
    #pragma unroll
    for (int i = 0; i < 2; ++i)
      #pragma unroll
      for (int j = 0; j < 4; ++j)
        acc[i][j] = __builtin_amdgcn_wmma_f32_16x16x32_bf16(
            false, a[i].v, false, b[j].v, (short)0, acc[i][j], false, false);
  };

  const int Kfull = K & ~(TK - 1);
  for (int k0 = 0; k0 < Kfull; k0 += TK) {
    stage(k0, false);
    __syncthreads();
    compute();
    __syncthreads();
  }
  if (Kfull < K) {
    stage(Kfull, true);
    __syncthreads();
    compute();
    __syncthreads();
  }

  // epilogue: C layout -> VGPR r: lanes 0-15 M=r, lanes 16-31 M=r+8; N = lane&15
  #pragma unroll
  for (int i = 0; i < 2; ++i) {
    #pragma unroll
    for (int j = 0; j < 4; ++j) {
      #pragma unroll
      for (int r = 0; r < 8; ++r) {
        int m = m0 + wm + i * 16 + r + halfsel * 8;
        int n = n0 + wn + j * 16 + l16;
        if (m < M && n < N) {
          float v = acc[i][j][r];
          if (bias) v += bias[n];
          if (act == 1) v = gelu_f(v);
          size_t off = (size_t)m * ldc + n;
          if (addC) v += C[off];
          C[off] = v;
        }
      }
    }
  }
}

// ---------------- cosine similarity ----------------
__global__ void sem_kernel(const float* __restrict__ center,
                           const float* __restrict__ cand,
                           float* __restrict__ sem) {
  int row = blockIdx.x * 8 + (threadIdx.x >> 5);
  int lane = threadIdx.x & 31;
  const float* c = center + (size_t)(row >> 9) * TDd;
  const float* e = cand + (size_t)row * TDd;
  float d = 0.f, cn = 0.f, en = 0.f;
  for (int i = lane; i < TDd; i += 32) {
    float a = c[i], b2 = e[i];
    d += a * b2; cn += a * a; en += b2 * b2;
  }
  for (int o = 16; o > 0; o >>= 1) {
    d += __shfl_down(d, o); cn += __shfl_down(cn, o); en += __shfl_down(en, o);
  }
  if (lane == 0) sem[row] = d / (sqrtf(cn) * sqrtf(en) + 1e-8f);
}

// ---------------- feats concat ----------------
__global__ void feats_kernel(const float* __restrict__ qc, const float* __restrict__ cand,
                             const float* __restrict__ edge_tbl, const int* __restrict__ et,
                             const float* __restrict__ year, const float* __restrict__ score,
                             const float* __restrict__ sem,
                             const float* __restrict__ tW, const float* __restrict__ tB,
                             const float* __restrict__ sW, const float* __restrict__ sB,
                             float* __restrict__ F) {
  int row = blockIdx.x;
  int b = row >> 9;
  float yr = year[row], sc = score[row], se = sem[row];
  int e = et[row]; e = e < 0 ? 0 : (e > 3 ? 3 : e);
  float* out = F + (size_t)row * 1281;
  for (int c = threadIdx.x; c < 1281; c += 256) {
    float v;
    if (c < 256)       v = qc[b * 256 + c];
    else if (c < 512)  v = cand[(size_t)row * 256 + (c - 256)];
    else if (c < 768)  v = edge_tbl[e * 256 + (c - 512)];
    else if (c < 1024) v = yr * tW[c - 768] + tB[c - 768];
    else if (c < 1280) v = sc * sW[c - 1024] + sB[c - 1024];
    else               v = se;
    out[c] = v;
  }
}

// ---------------- logits = h1 . w2 + b, masked ----------------
__global__ void logits_kernel(const float* __restrict__ H1, const float* __restrict__ w2,
                              const float* __restrict__ b2, const unsigned char* __restrict__ mask,
                              float* __restrict__ out) {
  int row = blockIdx.x * 8 + (threadIdx.x >> 5);
  int lane = threadIdx.x & 31;
  const float* h = H1 + (size_t)row * 256;
  float s = 0.f;
  for (int i = lane; i < 256; i += 32) s += h[i] * w2[i];
  for (int o = 16; o > 0; o >>= 1) s += __shfl_down(s, o);
  if (lane == 0) {
    float v = s + b2[0];
    out[row] = mask[row] ? v : NEGV;
  }
}

// ---------------- top-64 (selection, sorted desc, lowest idx on ties) ----------------
__global__ void topk_kernel(const float* __restrict__ logits,
                            float* __restrict__ topv, int* __restrict__ topi) {
  int b = blockIdx.x, tid = threadIdx.x;
  __shared__ float sv[Nn];
  __shared__ float rv[256];
  __shared__ int   ri[256];
  for (int i = tid; i < Nn; i += 256) sv[i] = logits[b * Nn + i];
  __syncthreads();
  for (int t = 0; t < Kk; ++t) {
    float v1 = sv[tid]; int i1 = tid;
    float v2 = sv[tid + 256]; int i2 = tid + 256;
    if (v2 > v1) { v1 = v2; i1 = i2; }
    rv[tid] = v1; ri[tid] = i1;
    __syncthreads();
    for (int s = 128; s > 0; s >>= 1) {
      if (tid < s) {
        float vo = rv[tid + s]; int io = ri[tid + s];
        if (vo > rv[tid] || (vo == rv[tid] && io < ri[tid])) { rv[tid] = vo; ri[tid] = io; }
      }
      __syncthreads();
    }
    if (tid == 0) {
      topv[b * Kk + t] = rv[0];
      topi[b * Kk + t] = ri[0];
      sv[ri[0]] = -3.0e38f;
    }
    __syncthreads();
  }
}

// ---------------- gather selected candidates ----------------
__global__ void gather_kernel(const int* __restrict__ topi, const float* __restrict__ cand_emb,
                              const int* __restrict__ et, const float* __restrict__ year,
                              const unsigned char* __restrict__ mask, const float* __restrict__ hop,
                              float* __restrict__ selemb, int* __restrict__ seledge,
                              float* __restrict__ selyear, float* __restrict__ selmask,
                              float* __restrict__ selhop) {
  int i = blockIdx.x;
  int b = i >> 6;
  int idx = topi[i];
  const float* src = cand_emb + (size_t)(b * Nn + idx) * TDd;
  float* dst = selemb + (size_t)i * TDd;
  for (int t = threadIdx.x; t < TDd; t += 256) dst[t] = src[t];
  if (threadIdx.x == 0) {
    int e = et[b * Nn + idx]; e = e < 0 ? 0 : (e > 3 ? 3 : e);
    seledge[i] = e;
    selyear[i] = year[b * Nn + idx];
    selmask[i] = mask[b * Nn + idx] ? 1.f : 0.f;
    selhop[2 * i]     = hop[(size_t)(b * Nn + idx) * 2];
    selhop[2 * i + 1] = hop[(size_t)(b * Nn + idx) * 2 + 1];
  }
}

// ---------------- struct features + add + LayerNorm -> tokens[1..64] ----------------
__global__ void structln_kernel(const float* __restrict__ text, const float* __restrict__ tok_edge,
                                const int* __restrict__ seledge, const float* __restrict__ selyear,
                                const float* __restrict__ topv, const float* __restrict__ selhop,
                                const float* __restrict__ tW, const float* __restrict__ tB,
                                const float* __restrict__ scW, const float* __restrict__ scB,
                                const float* __restrict__ hW, const float* __restrict__ hB,
                                const float* __restrict__ spB,
                                const float* __restrict__ g, const float* __restrict__ be,
                                float* __restrict__ tokens) {
  int row = blockIdx.x;
  int b = row >> 6, kk = row & 63;
  __shared__ float xs[Dd];
  __shared__ float red[256];
  int e = seledge[row];
  float yr = selyear[row], lg = topv[row];
  float h0 = selhop[2 * row], h1v = selhop[2 * row + 1];
  for (int d = threadIdx.x; d < Dd; d += 256) {
    float s = tok_edge[e * Dd + d] + yr * tW[d] + tB[d] + lg * scW[d] + scB[d]
            + h0 * hW[d] + h1v * hW[Dd + d] + hB[d] + spB[d];
    xs[d] = text[(size_t)row * Dd + d] + s;
  }
  __syncthreads();
  float s = 0.f;
  for (int d = threadIdx.x; d < Dd; d += 256) s += xs[d];
  red[threadIdx.x] = s; __syncthreads();
  for (int st = 128; st > 0; st >>= 1) { if (threadIdx.x < st) red[threadIdx.x] += red[threadIdx.x + st]; __syncthreads(); }
  float mean = red[0] / (float)Dd; __syncthreads();
  float v = 0.f;
  for (int d = threadIdx.x; d < Dd; d += 256) { float t = xs[d] - mean; v += t * t; }
  red[threadIdx.x] = v; __syncthreads();
  for (int st = 128; st > 0; st >>= 1) { if (threadIdx.x < st) red[threadIdx.x] += red[threadIdx.x + st]; __syncthreads(); }
  float rstd = rsqrtf(red[0] / (float)Dd + 1e-5f);
  float* out = tokens + (size_t)(b * Ss + 1 + kk) * Dd;
  for (int d = threadIdx.x; d < Dd; d += 256)
    out[d] = (xs[d] - mean) * rstd * g[d] + be[d];
}

// ---------------- generic LayerNorm (D=512) ----------------
__global__ void layernorm_kernel(const float* __restrict__ X, const float* __restrict__ g,
                                 const float* __restrict__ be, float* __restrict__ Y) {
  int row = blockIdx.x;
  const float* x = X + (size_t)row * Dd;
  __shared__ float red[256];
  float s = 0.f;
  for (int d = threadIdx.x; d < Dd; d += 256) s += x[d];
  red[threadIdx.x] = s; __syncthreads();
  for (int st = 128; st > 0; st >>= 1) { if (threadIdx.x < st) red[threadIdx.x] += red[threadIdx.x + st]; __syncthreads(); }
  float mean = red[0] / (float)Dd; __syncthreads();
  float v = 0.f;
  for (int d = threadIdx.x; d < Dd; d += 256) { float t = x[d] - mean; v += t * t; }
  red[threadIdx.x] = v; __syncthreads();
  for (int st = 128; st > 0; st >>= 1) { if (threadIdx.x < st) red[threadIdx.x] += red[threadIdx.x + st]; __syncthreads(); }
  float rstd = rsqrtf(red[0] / (float)Dd + 1e-5f);
  for (int d = threadIdx.x; d < Dd; d += 256)
    Y[(size_t)row * Dd + d] = (x[d] - mean) * rstd * g[d] + be[d];
}

// ---------------- attention (per (b,h); S=65, HD=64) ----------------
__global__ __launch_bounds__(128) void attn_kernel(
    const float* __restrict__ Q, const float* __restrict__ Kb, const float* __restrict__ V,
    const float* __restrict__ ab, const float* __restrict__ mc,
    const float* __restrict__ sbW, const float* __restrict__ sbB,
    const float* __restrict__ mbW, const float* __restrict__ mbB,
    const float* __restrict__ selmask, float* __restrict__ O) {
  int bh = blockIdx.x;
  int b = bh >> 3, h = bh & 7;
  __shared__ __align__(16) float ks[Ss][HDd];
  __shared__ __align__(16) float vs[Ss][HDd];
  __shared__ float qv[HDd];
  __shared__ float sc[Ss];
  __shared__ float stat;
  int tid = threadIdx.x;
#ifdef HAVE_ASYNC_LDS
  // CDNA5 async global->LDS staging (ASYNCcnt + s_wait_asynccnt)
  for (int i = tid * 4; i < Ss * HDd; i += 128 * 4) {
    int s = i >> 6, d = i & 63;
    size_t off = (size_t)(b * Ss + s) * Dd + h * HDd + d;
    __builtin_amdgcn_global_load_async_to_lds_b128((v4i*)&Kb[off], (v4i*)&ks[s][d], 0, 0);
    __builtin_amdgcn_global_load_async_to_lds_b128((v4i*)&V[off],  (v4i*)&vs[s][d], 0, 0);
  }
  __builtin_amdgcn_s_wait_asynccnt(0);
  __syncthreads();
#else
  for (int i = tid; i < Ss * HDd; i += 128) {
    int s = i >> 6, d = i & 63;
    size_t off = (size_t)(b * Ss + s) * Dd + h * HDd + d;
    ks[s][d] = Kb[off];
    vs[s][d] = V[off];
  }
  __syncthreads();
#endif
  for (int q = 0; q < Ss; ++q) {
    for (int d = tid; d < HDd; d += 128)
      qv[d] = Q[(size_t)(b * Ss + q) * Dd + h * HDd + d];
    __syncthreads();
    for (int j = tid; j < Ss; j += 128) {
      float dot = 0.f;
      #pragma unroll 8
      for (int d = 0; d < HDd; ++d) dot += qv[d] * ks[j][d];
      size_t bi = ((size_t)(b * Ss + q) * Ss + j);
      float bias = ab[bi * 3 + 0] * sbW[0 * Hh + h] + ab[bi * 3 + 1] * sbW[1 * Hh + h]
                 + ab[bi * 3 + 2] * sbW[2 * Hh + h] + sbB[h]
                 + mc[bi] * mbW[h] + mbB[h];
      float val = dot * SCALEV + bias;
      if (j > 0 && selmask[b * Kk + (j - 1)] == 0.f) val = NEGV;
      sc[j] = val;
    }
    __syncthreads();
    if (tid == 0) {
      float mx = -3.0e38f;
      for (int j = 0; j < Ss; ++j) mx = fmaxf(mx, sc[j]);
      float sum = 0.f;
      for (int j = 0; j < Ss; ++j) { float e = expf(sc[j] - mx); sc[j] = e; sum += e; }
      stat = 1.0f / sum;
    }
    __syncthreads();
    for (int d = tid; d < HDd; d += 128) {
      float acc = 0.f;
      for (int j = 0; j < Ss; ++j) acc += sc[j] * vs[j][d];
      O[(size_t)(b * Ss + q) * Dd + h * HDd + d] = acc * stat;
    }
    __syncthreads();
  }
}

// ---------------- masked mean pool + concat ----------------
__global__ void pool_kernel(const float* __restrict__ tokens, const float* __restrict__ selmask,
                            float* __restrict__ meanb, float* __restrict__ cc) {
  int b = blockIdx.x;
  float cnt = 0.f;
  for (int kk = 0; kk < Kk; ++kk) cnt += selmask[b * Kk + kk];
  float denom = fmaxf(cnt, 1.0f);
  for (int d = threadIdx.x; d < Dd; d += 256) {
    float s = 0.f;
    for (int kk = 0; kk < Kk; ++kk)
      s += selmask[b * Kk + kk] * tokens[(size_t)(b * Ss + 1 + kk) * Dd + d];
    float mean = s / denom;
    meanb[b * Dd + d] = mean;
    cc[(size_t)b * 2 * Dd + d]      = tokens[(size_t)(b * Ss) * Dd + d];
    cc[(size_t)b * 2 * Dd + Dd + d] = mean;
  }
}

// ---------------- sigmoid gate combine ----------------
__global__ void combine_kernel(const float* __restrict__ G, const float* __restrict__ tokens,
                               const float* __restrict__ meanb, float* __restrict__ pooled) {
  int i = blockIdx.x * 256 + threadIdx.x;
  int b = i >> 9, d = i & 511;
  float g = 1.0f / (1.0f + expf(-G[i]));
  pooled[i] = g * tokens[(size_t)(b * Ss) * Dd + d] + (1.0f - g) * meanb[i];
}

// ---------------- launch ----------------
extern "C" void kernel_launch(void* const* d_in, const int* in_sizes, int n_in,
                              void* d_out, int out_size, void* d_ws, size_t ws_size,
                              hipStream_t stream) {
  (void)in_sizes; (void)n_in; (void)out_size; (void)ws_size;
  auto FP = [&](int i) { return (const float*)d_in[i]; };

  // inputs (setup_inputs order)
  const float* center_text = FP(0);
  const float* cand_emb    = FP(1);
  const unsigned char* cmask = (const unsigned char*)d_in[2];
  const int*   etypes      = (const int*)d_in[3];
  const float* year        = FP(4);
  const float* cache       = FP(5);
  const float* hop         = FP(6);
  const float* metadata    = FP(7);
  const float* ab          = FP(8);
  const float* mc          = FP(9);

  // params: pytree flatten (alphabetical dict keys, tuples/lists in order)
  const int P = 10;
  const float* centerW = FP(P + 0);  const float* centerB = FP(P + 1);
  const float* gateW   = FP(P + 2);  const float* gateB   = FP(P + 3);
  const int LBASE = P + 4;                     // layers[0..3], 20 leaves each
  const float* metaW   = FP(P + 84); const float* metaB   = FP(P + 85);
  const float* outW    = FP(P + 86); const float* outB    = FP(P + 87);
  const float* onG     = FP(P + 88); const float* onB     = FP(P + 89);
  const float* selcW   = FP(P + 90); const float* selcB   = FP(P + 91);
  const float* seledgeT= FP(P + 92);
  const float* f1W     = FP(P + 93); const float* f1B     = FP(P + 94);
  const float* f2W     = FP(P + 95); const float* f2B     = FP(P + 96);
  const float* selqW   = FP(P + 97); const float* selqB   = FP(P + 98);
  const float* selsW   = FP(P + 99); const float* selsB   = FP(P + 100);
  const float* seltW   = FP(P + 101);const float* seltB   = FP(P + 102);
  const float* tokEdge = FP(P + 103);
  const float* hopW    = FP(P + 104);const float* hopB    = FP(P + 105);
  const float* tnG     = FP(P + 106);const float* tnB     = FP(P + 107);
  const float* scoreW  = FP(P + 108);const float* scoreB  = FP(P + 109);
  /* tok_spec.W unused (input is zeros) */    const float* specB = FP(P + 111);
  const float* textW   = FP(P + 112);const float* textB   = FP(P + 113);
  const float* timeW   = FP(P + 114);const float* timeB   = FP(P + 115);

  // workspace layout (floats)
  float* ws = (float*)d_ws;
  size_t off = 0;
  auto AL = [&](size_t n) { float* p = ws + off; off += n; return p; };
  float* qc      = AL(128 * 256);
  float* semb    = AL(65536);
  float* logitsb = AL(65536);
  float* topv    = AL(8192);
  int*   topi    = (int*)AL(8192);
  int*   seledge = (int*)AL(8192);
  float* selyear = AL(8192);
  float* selmask = AL(8192);
  float* selhop  = AL(16384);
  float* meanb   = AL(65536);
  float* cc      = AL(131072);
  float* gbuf    = AL(65536);
  float* pooled  = AL(65536);
  float* nbuf    = AL(65536);
  // big region A: cand (16.78M) later reused for selemb+text
  float* cand    = AL((size_t)65536 * 256);
  float* selemb  = cand;
  float* text    = cand + (size_t)8192 * 768;
  // big region B: feats (83.95M) later reused for transformer buffers
  float* feats   = AL((size_t)65536 * 1281);
  float* tokens  = feats;
  float* xbuf    = tokens + (size_t)8320 * 512;
  float* qbuf    = xbuf   + (size_t)8320 * 512;
  float* kbuf    = qbuf   + (size_t)8320 * 512;
  float* vbuf    = kbuf   + (size_t)8320 * 512;
  float* obuf    = vbuf   + (size_t)8320 * 512;
  float* ybuf    = obuf   + (size_t)8320 * 512;
  float* ffbuf   = ybuf   + (size_t)8320 * 512;
  float* h1      = AL((size_t)65536 * 256);

  auto GEMM = [&](const float* A, int lda, const float* W, const float* bias,
                  float* C, int ldc, int M, int N, int K, int act, int addC) {
    dim3 grid(N / TN, M / TM);
    wmma_gemm_kernel<<<grid, 256, 0, stream>>>(A, lda, W, bias, C, ldc, M, N, K, act, addC);
  };

  // ---- selection stage ----
  GEMM(center_text, TDd, selqW, selqB, qc, HSs, 128, HSs, TDd, 0, 0);
  GEMM(cand_emb, TDd, selcW, selcB, cand, HSs, 65536, HSs, TDd, 0, 0);
  sem_kernel<<<65536 / 8, 256, 0, stream>>>(center_text, cand_emb, semb);
  feats_kernel<<<65536, 256, 0, stream>>>(qc, cand, seledgeT, etypes, year, cache, semb,
                                          seltW, seltB, selsW, selsB, feats);
  GEMM(feats, 1281, f1W, f1B, h1, HSs, 65536, HSs, 1281, 1, 0);
  logits_kernel<<<65536 / 8, 256, 0, stream>>>(h1, f2W, f2B, cmask, logitsb);
  topk_kernel<<<Bb, 256, 0, stream>>>(logitsb, topv, topi);
  gather_kernel<<<Bb * Kk, 256, 0, stream>>>(topi, cand_emb, etypes, year, cmask, hop,
                                             selemb, seledge, selyear, selmask, selhop);

  // ---- token build ----
  GEMM(selemb, TDd, textW, textB, text, Dd, Bb * Kk, Dd, TDd, 0, 0);
  structln_kernel<<<Bb * Kk, 256, 0, stream>>>(text, tokEdge, seledge, selyear, topv, selhop,
                                               timeW, timeB, scoreW, scoreB, hopW, hopB, specB,
                                               tnG, tnB, tokens);
  GEMM(center_text, TDd, centerW, centerB, tokens, Ss * Dd, Bb, Dd, TDd, 0, 0);
  GEMM(metadata, MDd, metaW, metaB, tokens, Ss * Dd, Bb, Dd, MDd, 0, 1);

  // ---- transformer layers ----
  const int Mtok = Bb * Ss;   // 8320 = 65 * 128
  for (int l = 0; l < NLl; ++l) {
    int lb = LBASE + l * 20;
    const float* lf1W = FP(lb + 0);  const float* lf1B = FP(lb + 1);
    const float* lf2W = FP(lb + 2);  const float* lf2B = FP(lb + 3);
    const float* lkW  = FP(lb + 4);  const float* lkB  = FP(lb + 5);
    const float* lmbW = FP(lb + 6);  const float* lmbB = FP(lb + 7);
    const float* ln1G = FP(lb + 8);  const float* ln1B = FP(lb + 9);
    const float* ln2G = FP(lb + 10); const float* ln2B = FP(lb + 11);
    const float* loW  = FP(lb + 12); const float* loB  = FP(lb + 13);
    const float* lqW  = FP(lb + 14); const float* lqB  = FP(lb + 15);
    const float* lsbW = FP(lb + 16); const float* lsbB = FP(lb + 17);
    const float* lvW  = FP(lb + 18); const float* lvB  = FP(lb + 19);

    layernorm_kernel<<<Mtok, 256, 0, stream>>>(tokens, ln1G, ln1B, xbuf);
    GEMM(xbuf, Dd, lqW, lqB, qbuf, Dd, Mtok, Dd, Dd, 0, 0);
    GEMM(xbuf, Dd, lkW, lkB, kbuf, Dd, Mtok, Dd, Dd, 0, 0);
    GEMM(xbuf, Dd, lvW, lvB, vbuf, Dd, Mtok, Dd, Dd, 0, 0);
    attn_kernel<<<Bb * Hh, 128, 0, stream>>>(qbuf, kbuf, vbuf, ab, mc,
                                             lsbW, lsbB, lmbW, lmbB, selmask, obuf);
    GEMM(obuf, Dd, loW, loB, tokens, Dd, Mtok, Dd, Dd, 0, 1);
    layernorm_kernel<<<Mtok, 256, 0, stream>>>(tokens, ln2G, ln2B, ybuf);
    GEMM(ybuf, Dd, lf1W, lf1B, ffbuf, FFf, Mtok, FFf, Dd, 1, 0);
    GEMM(ffbuf, FFf, lf2W, lf2B, tokens, Dd, Mtok, Dd, FFf, 0, 1);
  }

  // ---- pooling / gate / output ----
  pool_kernel<<<Bb, 256, 0, stream>>>(tokens, selmask, meanb, cc);
  GEMM(cc, 2 * Dd, gateW, gateB, gbuf, Dd, Bb, Dd, 2 * Dd, 0, 0);
  combine_kernel<<<(Bb * Dd) / 256, 256, 0, stream>>>(gbuf, tokens, meanb, pooled);
  layernorm_kernel<<<Bb, 256, 0, stream>>>(pooled, onG, onB, nbuf);
  GEMM(nbuf, Dd, outW, outB, (float*)d_out, Dd, Bb, Dd, Dd, 0, 0);
}
